// SparseAttention_38328288150104
// MI455X (gfx1250) — matrix operand order repaired and checked by reading, not compiled, not run on hardware
//
#include <hip/hip_runtime.h>

// ---------------------------------------------------------------------------
// Types / WMMA helpers (gfx1250, wave32)
// ---------------------------------------------------------------------------
typedef __bf16 bf16_t;
typedef __attribute__((ext_vector_type(16))) __bf16 v16bf;
typedef __attribute__((ext_vector_type(8)))  __bf16 v8bf;
typedef __attribute__((ext_vector_type(8)))  float  v8f;

#define DEV __device__ __forceinline__

#define B_  16
#define N_  1024
#define C_  768
#define H_  12
#define D_  64
#define K3_ 2304   // 3*C

DEV v16bf frag_cat(v8bf lo, v8bf hi) {
    v16bf r;
#pragma unroll
    for (int i = 0; i < 8; ++i) { r[i] = lo[i]; r[8 + i] = hi[i]; }
    return r;
}

// A-matrix fragment (16x32 bf16, row-major source with leading dim `ld`).
// ISA layout: lanes0-15 row=lane, K = {0..7,16..23}; lanes16-31 row=lane-16,
// K = {8..15,24..31}.  Two 16B loads per lane.
DEV v16bf load_a_bf(const bf16_t* base, int ld, int lane) {
    const int row = lane & 15;
    const int kh  = (lane >> 4) << 3;           // 0 or 8
    const bf16_t* p = base + (size_t)row * ld + kh;
    return frag_cat(*(const v8bf*)p, *(const v8bf*)(p + 16));
}

// B-matrix fragment (32x16 bf16) from COLUMN-major storage: element (k, n) at
// base[n*ld + k].  ISA layout: lanes0-15 col=lane, K = {0..7,16..23};
// lanes16-31 col=lane-16, K = {8..15,24..31}.  Two 16B loads per lane.
DEV v16bf load_b_cm(const bf16_t* base, int ld, int lane) {
    const int col = lane & 15;
    const int kb  = (lane >> 4) << 3;           // 0 or 8
    const bf16_t* p = base + (size_t)col * ld + kb;
    return frag_cat(*(const v8bf*)p, *(const v8bf*)(p + 16));
}

DEV v8f wmma_bf(v16bf a, v16bf b, v8f c) {
    return __builtin_amdgcn_wmma_f32_16x16x32_bf16(
        /*neg_a=*/false, a, /*neg_b=*/false, b,
        /*c_mod=*/(short)0, c, /*reuse_a=*/false, /*reuse_b=*/false);
}

// ---------------------------------------------------------------------------
// Prep: W (K x N, fp32 row-major) -> Wt (N x K, bf16) so B-frags are contiguous
// ---------------------------------------------------------------------------
__global__ void transpose_bf16(const float* __restrict__ W, bf16_t* __restrict__ Wt,
                               int K, int N) {
    int idx = blockIdx.x * 256 + threadIdx.x;   // output-coalesced
    if (idx >= K * N) return;
    int n = idx / K;
    int k = idx - n * K;
    Wt[idx] = (bf16_t)W[(size_t)k * N + n];
}

// ---------------------------------------------------------------------------
// GEMM: per wave a 16x64 output tile (4 fp32 accumulators), K = 768.
// MODE 0: A = x (fp32, convert on the fly) -> QKV epilogue (bias, zeta,
//         scatter q/k row-major and v transposed, all bf16).
// MODE 1: A = attn output (bf16)           -> out = A*Wproj + bias (fp32).
// ---------------------------------------------------------------------------
template <int MODE>
__global__ __launch_bounds__(256)
void gemm_kernel(const float* __restrict__ Xf, const bf16_t* __restrict__ Xb,
                 const bf16_t* __restrict__ Wt, const float* __restrict__ bias,
                 const float* __restrict__ zeta,
                 bf16_t* __restrict__ qbuf, bf16_t* __restrict__ kbuf,
                 bf16_t* __restrict__ vtbuf, float* __restrict__ out) {
    const int tid  = threadIdx.x;
    const int wave = tid >> 5;
    const int lane = tid & 31;
    const int m0   = blockIdx.y * 128 + wave * 16;   // 8 waves stack 16-row tiles
    const int cb   = blockIdx.x * 64;                // 64 output columns / block

    const int row = lane & 15;
    const int kh8 = (lane >> 4) << 3;

    v8f acc[4];
#pragma unroll
    for (int j = 0; j < 4; ++j)
#pragma unroll
        for (int g = 0; g < 8; ++g) acc[j][g] = 0.0f;

    for (int kk = 0; kk < C_; kk += 32) {
        v16bf a;
        if (MODE == 0) {
            const float* ap = Xf + (size_t)(m0 + row) * C_ + kk + kh8;
            __builtin_prefetch(ap + C_, 0, 0);       // next-row stream hint
            float4 f0 = *(const float4*)(ap);
            float4 f1 = *(const float4*)(ap + 4);
            float4 f2 = *(const float4*)(ap + 16);
            float4 f3 = *(const float4*)(ap + 20);
            a[0]=(bf16_t)f0.x; a[1]=(bf16_t)f0.y; a[2]=(bf16_t)f0.z; a[3]=(bf16_t)f0.w;
            a[4]=(bf16_t)f1.x; a[5]=(bf16_t)f1.y; a[6]=(bf16_t)f1.z; a[7]=(bf16_t)f1.w;
            a[8]=(bf16_t)f2.x; a[9]=(bf16_t)f2.y; a[10]=(bf16_t)f2.z; a[11]=(bf16_t)f2.w;
            a[12]=(bf16_t)f3.x; a[13]=(bf16_t)f3.y; a[14]=(bf16_t)f3.z; a[15]=(bf16_t)f3.w;
        } else {
            a = load_a_bf(Xb + (size_t)m0 * C_ + kk, C_, lane);
        }
#pragma unroll
        for (int j = 0; j < 4; ++j) {
            v16bf bfr = load_b_cm(Wt + (size_t)(cb + j * 16) * C_ + kk, C_, lane);
            acc[j] = wmma_bf(a, bfr, acc[j]);
        }
    }

    if (MODE == 0) {
        const int which = cb / C_;                   // 0=q 1=k 2=v (uniform/block)
        const int hh    = (cb % C_) / D_;            // head      (uniform/block)
#pragma unroll
        for (int j = 0; j < 4; ++j) {
            const int dd = j * 16 + (lane & 15);     // dim within head (cb%64==0)
            const float bv = bias[cb + dd];
            const float zv = zeta[hh * D_ + dd];
#pragma unroll
            for (int g = 0; g < 8; ++g) {
                const int r  = m0 + ((lane < 16) ? g : 8 + g);
                const int bb = r >> 10;              // batch
                const int n  = r & 1023;             // token
                const size_t bh = (size_t)bb * H_ + hh;
                const bf16_t v = (bf16_t)((acc[j][g] + bv) * zv);
                if (which == 0)      qbuf [(bh * N_ + n) * D_ + dd]  = v;
                else if (which == 1) kbuf [(bh * N_ + n) * D_ + dd]  = v;
                else                 vtbuf[(bh * D_ + dd) * N_ + n]  = v;  // transposed
            }
        }
    } else {
#pragma unroll
        for (int j = 0; j < 4; ++j) {
            const int c  = cb + j * 16 + (lane & 15);
            const float bv = bias[c];
#pragma unroll
            for (int g = 0; g < 8; ++g) {
                const int r = m0 + ((lane < 16) ? g : 8 + g);
                out[(size_t)r * C_ + c] = acc[j][g] + bv;
            }
        }
    }
}

// ---------------------------------------------------------------------------
// Flash attention: 8 waves/block, wave = 16 queries, key blocks of 64,
// online softmax, bf16 WMMA for QK^T and PV.  P transits per-wave LDS
// (stride 72 bf16 to avoid bank conflicts) for C-layout -> A-layout.
// ---------------------------------------------------------------------------
__global__ __launch_bounds__(256)
void attn_kernel(const bf16_t* __restrict__ qb, const bf16_t* __restrict__ kb,
                 const bf16_t* __restrict__ vtb, const unsigned char* __restrict__ mask,
                 bf16_t* __restrict__ aout) {
    __shared__ bf16_t plds[8 * 16 * 72];
    const int tid  = threadIdx.x;
    const int wave = tid >> 5;
    const int lane = tid & 31;
    const int bh   = blockIdx.y;                 // 0..191
    const int b    = bh / H_;
    const int h    = bh - b * H_;
    const int q0   = blockIdx.x * 128 + wave * 16;

    const bf16_t* qp = qb  + ((size_t)bh * N_ + q0) * D_;
    const bf16_t* kp = kb  + (size_t)bh * N_ * D_;
    const bf16_t* vp = vtb + (size_t)bh * D_ * N_;
    const unsigned char* mp = mask + (size_t)b * N_;
    bf16_t* pl = plds + wave * 16 * 72;

    // Q fragments, resident for the whole K-loop
    const v16bf qa0 = load_a_bf(qp + 0,  D_, lane);
    const v16bf qa1 = load_a_bf(qp + 32, D_, lane);

    v8f o[4];
#pragma unroll
    for (int j = 0; j < 4; ++j)
#pragma unroll
        for (int g = 0; g < 8; ++g) o[j][g] = 0.0f;
    float mrow[8], lrow[8];
#pragma unroll
    for (int g = 0; g < 8; ++g) { mrow[g] = -1e30f; lrow[g] = 0.0f; }

    const float scale = 0.125f;                  // d^-0.5, d=64

    for (int k0 = 0; k0 < N_; k0 += 64) {
        // ---- S = scale * Q K^T, masked --------------------------------
        v8f s[4];
#pragma unroll
        for (int j = 0; j < 4; ++j) {
            const bf16_t* kbase = kp + (size_t)(k0 + j * 16) * D_;
            v16bf kb0 = load_b_cm(kbase + 0,  D_, lane);   // d 0..31
            v16bf kb1 = load_b_cm(kbase + 32, D_, lane);   // d 32..63
            v8f z;
#pragma unroll
            for (int g = 0; g < 8; ++g) z[g] = 0.0f;
            z    = wmma_bf(qa0, kb0, z);
            s[j] = wmma_bf(qa1, kb1, z);
            const float madd = mp[k0 + j * 16 + (lane & 15)] ? -1e30f : 0.0f;
#pragma unroll
            for (int g = 0; g < 8; ++g) s[j][g] = s[j][g] * scale + madd;
        }
        // ---- online softmax (rows split 8/8 across lane halves) -------
#pragma unroll
        for (int g = 0; g < 8; ++g) {
            float rm = fmaxf(fmaxf(s[0][g], s[1][g]), fmaxf(s[2][g], s[3][g]));
            rm = fmaxf(rm, __shfl_xor(rm, 1, 16));
            rm = fmaxf(rm, __shfl_xor(rm, 2, 16));
            rm = fmaxf(rm, __shfl_xor(rm, 4, 16));
            rm = fmaxf(rm, __shfl_xor(rm, 8, 16));
            const float mn    = fmaxf(mrow[g], rm);
            const float alpha = __expf(mrow[g] - mn);
            mrow[g] = mn;
            const int prow = (lane < 16) ? g : 8 + g;
            float ps = 0.0f;
#pragma unroll
            for (int j = 0; j < 4; ++j) {
                const float p = __expf(s[j][g] - mn);
                ps += p;
                pl[prow * 72 + j * 16 + (lane & 15)] = (bf16_t)p;
            }
            ps += __shfl_xor(ps, 1, 16);
            ps += __shfl_xor(ps, 2, 16);
            ps += __shfl_xor(ps, 4, 16);
            ps += __shfl_xor(ps, 8, 16);
            lrow[g] = lrow[g] * alpha + ps;
#pragma unroll
            for (int j = 0; j < 4; ++j) o[j][g] *= alpha;
        }
        // ---- O += P V  (P re-read from LDS in A-layout) ---------------
        {
            const int prow2 = lane & 15;
            const int kh    = (lane >> 4) << 3;
            const bf16_t* pp = pl + prow2 * 72 + kh;
            v16bf pa0 = frag_cat(*(const v8bf*)pp,        *(const v8bf*)(pp + 16));
            v16bf pa1 = frag_cat(*(const v8bf*)(pp + 32), *(const v8bf*)(pp + 48));
#pragma unroll
            for (int j = 0; j < 4; ++j) {
                const bf16_t* vbase = vp + (size_t)(j * 16) * N_ + k0;
                v16bf vb0 = load_b_cm(vbase,      N_, lane);   // keys k0..k0+31
                o[j] = wmma_bf(pa0, vb0, o[j]);
                v16bf vb1 = load_b_cm(vbase + 32, N_, lane);   // keys +32..+63
                o[j] = wmma_bf(pa1, vb1, o[j]);
            }
        }
    }

    // ---- normalize and store to (B, N, C) layout (bf16) ---------------
#pragma unroll
    for (int g = 0; g < 8; ++g) {
        const float inv = 1.0f / lrow[g];
        const int   r   = q0 + ((lane < 16) ? g : 8 + g);
        const size_t base = ((size_t)(b * N_ + r)) * C_ + h * D_ + (lane & 15);
#pragma unroll
        for (int j = 0; j < 4; ++j)
            aout[base + j * 16] = (bf16_t)(o[j][g] * inv);
    }
}

// ---------------------------------------------------------------------------
// Launch
// ---------------------------------------------------------------------------
extern "C" void kernel_launch(void* const* d_in, const int* in_sizes, int n_in,
                              void* d_out, int out_size, void* d_ws, size_t ws_size,
                              hipStream_t stream) {
    const float* x     = (const float*)d_in[0];          // (16,1024,768)
    const float* Wqkv  = (const float*)d_in[1];          // (768,2304)
    const float* bqkv  = (const float*)d_in[2];          // (2304,)
    const float* zeta  = (const float*)d_in[3];          // (H*d,) = 768
    const float* Wproj = (const float*)d_in[4];          // (768,768)
    const float* bproj = (const float*)d_in[5];          // (768,)
    const unsigned char* mask = (const unsigned char*)d_in[6]; // (16,1024) bool
    float* out = (float*)d_out;

    // workspace layout (bf16): transposed weights + q/k/vT + attn output
    bf16_t* wt_qkv  = (bf16_t*)d_ws;                         // 2304*768
    bf16_t* wt_proj = wt_qkv  + (size_t)K3_ * C_;            // 768*768
    bf16_t* qbuf    = wt_proj + (size_t)C_  * C_;            // 192*1024*64
    bf16_t* kbuf    = qbuf    + (size_t)B_ * H_ * N_ * D_;
    bf16_t* vtbuf   = kbuf    + (size_t)B_ * H_ * N_ * D_;
    bf16_t* attn    = vtbuf   + (size_t)B_ * H_ * N_ * D_;   // 16384*768

    transpose_bf16<<<(C_ * K3_ + 255) / 256, 256, 0, stream>>>(Wqkv, wt_qkv, C_, K3_);
    transpose_bf16<<<(C_ * C_  + 255) / 256, 256, 0, stream>>>(Wproj, wt_proj, C_, C_);

    // QKV projection: grid = (2304/64 col tiles, 16384/128 row tiles)
    gemm_kernel<0><<<dim3(K3_ / 64, (B_ * N_) / 128), 256, 0, stream>>>(
        x, nullptr, wt_qkv, bqkv, zeta, qbuf, kbuf, vtbuf, nullptr);

    // Flash attention: grid = (1024/128 query tiles, B*H heads)
    attn_kernel<<<dim3(N_ / 128, B_ * H_), 256, 0, stream>>>(
        qbuf, kbuf, vtbuf, mask, attn);

    // Output projection
    gemm_kernel<1><<<dim3(C_ / 64, (B_ * N_) / 128), 256, 0, stream>>>(
        nullptr, attn, wt_proj, bproj, nullptr, nullptr, nullptr, nullptr, out);
}
